// DGCNN_semseg_11450382811597
// MI455X (gfx1250) — matrix operand order repaired and checked by reading, not compiled, not run on hardware
//
#include <hip/hip_runtime.h>
#include <hip/hip_bf16.h>

typedef __attribute__((ext_vector_type(16))) _Float16 v16h;
typedef __attribute__((ext_vector_type(8)))  float    v8f;

#define NPTS 4096
#define KNB  20
#define NCOLS (NPTS * KNB)
#define BATCH 8

struct __align__(16) U128 { unsigned x, y, z, w; };

__device__ inline v8f vzero8() {
  v8f z = {0.f, 0.f, 0.f, 0.f, 0.f, 0.f, 0.f, 0.f};
  return z;
}

__device__ inline v8f wmma_acc(v16h a, v16h b, v8f c) {
  return __builtin_amdgcn_wmma_f32_16x16x32_f16(false, a, false, b, (short)0, c,
                                                false, false);
}

// fp32-accurate product via hi/lo f16 split: Ah*Bh + Ah*Bl + Al*Bh
__device__ inline v8f wmma3(v16h ah, v16h al, v16h bh, v16h bl, v8f acc) {
  acc = wmma_acc(al, bh, acc);
  acc = wmma_acc(ah, bl, acc);
  acc = wmma_acc(ah, bh, acc);
  return acc;
}

__device__ inline void splitf(float f, _Float16& h, _Float16& l) {
  _Float16 hh = (_Float16)f;
  h = hh;
  l = (_Float16)(f - (float)hh);
}

__device__ inline unsigned packh(_Float16 a, _Float16 b) {
  return ((unsigned)__builtin_bit_cast(unsigned short, b) << 16) |
         (unsigned)__builtin_bit_cast(unsigned short, a);
}

__device__ inline float bnlrelu(float y, float g, float b) {
  y = y * (g * rsqrtf(1.f + 1e-5f)) + b;
  return y > 0.f ? y : 0.2f * y;
}

// ---------------------------------------------------------------- small utils
__global__ void k_build_x0(const float* __restrict__ coords,
                           const float* __restrict__ feats,
                           float* __restrict__ x0) {
  int t = blockIdx.x * blockDim.x + threadIdx.x;
  if (t >= BATCH * NPTS) return;
  int b = t / NPTS, n = t % NPTS;
  float* o = x0 + (size_t)b * 6 * NPTS;
  const float* cb = coords + ((size_t)b * NPTS + n) * 3;
  const float* fb = feats + (size_t)b * 3 * NPTS;
#pragma unroll
  for (int c = 0; c < 3; ++c) {
    o[c * NPTS + n] = cb[c];
    o[(3 + c) * NPTS + n] = fb[c * NPTS + n];
  }
}

__global__ void k_sq(const float* __restrict__ x, int C, int bstride,
                     float* __restrict__ sq) {
  int t = blockIdx.x * blockDim.x + threadIdx.x;
  if (t >= BATCH * NPTS) return;
  int b = t / NPTS, n = t % NPTS;
  const float* xb = x + (size_t)b * bstride;
  float s = 0.f;
  for (int c = 0; c < C; ++c) {
    float v = xb[c * NPTS + n];
    s = fmaf(v, v, s);
  }
  sq[t] = s;
}

// ---------------------------------------------------------------- brute kNN
template <int CH>
__global__ void k_knn(const float* __restrict__ x, int bstride,
                      const float* __restrict__ sq, int* __restrict__ idx_out) {
  const int TILE = 64;
  __shared__ float sx[CH * TILE];
  __shared__ float ssq[TILE];
  int b = blockIdx.y;
  int q = blockIdx.x * blockDim.x + threadIdx.x;
  const float* xb = x + (size_t)b * bstride;
  const float* sqb = sq + (size_t)b * NPTS;
  float xq[CH];
#pragma unroll
  for (int c = 0; c < CH; ++c) xq[c] = xb[c * NPTS + q];
  float sqq = sqb[q];
  float best[KNB];
  int bidx[KNB];
#pragma unroll
  for (int i = 0; i < KNB; ++i) {
    best[i] = -3.0e38f;
    bidx[i] = 0;
  }
  for (int m0 = 0; m0 < NPTS; m0 += TILE) {
    __syncthreads();
    for (int t = threadIdx.x; t < CH * TILE; t += blockDim.x) {
      int c = t / TILE, mm = t % TILE;
      sx[t] = xb[c * NPTS + m0 + mm];
    }
    for (int t = threadIdx.x; t < TILE; t += blockDim.x) ssq[t] = sqb[m0 + t];
    __syncthreads();
    for (int mm = 0; mm < TILE; ++mm) {
      float dot = 0.f;
#pragma unroll
      for (int c = 0; c < CH; ++c) dot = fmaf(xq[c], sx[c * TILE + mm], dot);
      float nd = 2.f * dot - sqq - ssq[mm];  // same formula as reference
      if (nd > best[KNB - 1]) {
        float cd = nd;
        int ci = m0 + mm;
#pragma unroll
        for (int i = 0; i < KNB; ++i) {
          if (cd > best[i]) {  // strict >: stable ties, lower index first
            float td = best[i];
            int ti = bidx[i];
            best[i] = cd; bidx[i] = ci;
            cd = td; ci = ti;
          }
        }
      }
    }
  }
  int* ob = idx_out + ((size_t)b * NPTS + q) * KNB;
#pragma unroll
  for (int i = 0; i < KNB; ++i) ob[i] = bidx[i];
}

// ------------------------------------------------ shared A-tile staging (LDS)
// LDS layout: 128 entries (mt 0..3 x lane 0..31), stride 20 dwords (80B):
//   u32 [0..7]  = 16 hi-halfs in WMMA v16h register order for that lane
//   u32 [8..15] = 16 lo-halfs
//   u32 [16..19]= pad (bank spread, keeps 16B alignment)
// Reading an A operand is then 4x ds_load_b128 with no repacking.
#define SA_HALFS (128 * 40)

__device__ inline void fill_sA(_Float16* sA, const float* __restrict__ W,
                               int Mtot, int Ktot, int mrow0, int k0) {
  unsigned* sw = (unsigned*)sA;
  for (int t = threadIdx.x; t < 64 * 16; t += blockDim.x) {
    int mm = t >> 4;           // row in 64-row group
    int kp = (t & 15) << 1;    // even k within 32-wide tile (pair kp, kp+1)
    int m = mrow0 + mm;
    int k = k0 + kp;
    const float* row = W + (size_t)m * Ktot;
    float w0 = (m < Mtot && k < Ktot) ? row[k] : 0.f;
    float w1 = (m < Mtot && (k + 1) < Ktot) ? row[k + 1] : 0.f;
    _Float16 h0, l0, h1, l1;
    splitf(w0, h0, l0);
    splitf(w1, h1, l1);
    // contiguous A-layout: lane half from k bit4, j from k bits {3:0}
    int lane = (mm & 15) | (kp & 16);
    int entry = ((mm >> 4) << 5) | lane;
    int j0 = kp & 15;
    sw[entry * 20 + (j0 >> 1)] = packh(h0, h1);
    sw[entry * 20 + 8 + (j0 >> 1)] = packh(l0, l1);
  }
}

__device__ inline void read_A(const _Float16* sA, int mt, int lane, v16h& ah,
                              v16h& al) {
  const U128* p = (const U128*)(sA + (size_t)(((mt << 5) | lane) * 40));
  union { U128 q[2]; v16h v; } a, b;
  a.q[0] = p[0];
  a.q[1] = p[1];
  b.q[0] = p[2];
  b.q[1] = p[3];
  ah = a.v;
  al = b.v;
}

// ---------------------------------------------------------------- edge conv
// out[64, NCOLS] = lrelu(bn(W[64,2C] * edge_feat[2C, NCOLS]))
__global__ void k_edgeconv(const float* __restrict__ W, int Cin,
                           const float* __restrict__ xb,
                           const int* __restrict__ ib,
                           const float* __restrict__ g,
                           const float* __restrict__ bb,
                           float* __restrict__ out) {
  __shared__ __align__(16) _Float16 sA[SA_HALFS];
  int lane = threadIdx.x & 31;
  int wid = threadIdx.x >> 5;
  int colt = blockIdx.x * (blockDim.x >> 5) + wid;
  int col = colt * 16 + (lane & 15);
  int half = lane >> 4;
  int khalf = half << 4;
  int n = col / KNB;
  int nb = ib[col];
  int Ktot = 2 * Cin;
  int Ktiles = (Ktot + 31) >> 5;
  v8f acc[4];
  acc[0] = vzero8(); acc[1] = vzero8(); acc[2] = vzero8(); acc[3] = vzero8();
  for (int kt = 0; kt < Ktiles; ++kt) {
    __syncthreads();
    fill_sA(sA, W, 64, Ktot, 0, kt * 32);
    __syncthreads();
    v16h bh, bl;
#pragma unroll
    for (int j = 0; j < 16; ++j) {
      int k = kt * 32 + khalf + j;
      float f = 0.f;
      if (k < Cin)
        f = xb[k * NPTS + nb] - xb[k * NPTS + n];   // neighbor - center
      else if (k < Ktot)
        f = xb[(k - Cin) * NPTS + n];               // center
      _Float16 h, l;
      splitf(f, h, l);
      bh[j] = h; bl[j] = l;
    }
#pragma unroll
    for (int mt = 0; mt < 4; ++mt) {
      v16h ah, al;
      read_A(sA, mt, lane, ah, al);
      acc[mt] = wmma3(ah, al, bh, bl, acc[mt]);
    }
  }
#pragma unroll
  for (int mt = 0; mt < 4; ++mt) {
#pragma unroll
    for (int v = 0; v < 8; ++v) {
      int o = mt * 16 + half * 8 + v;
      out[(size_t)o * NCOLS + col] = bnlrelu(acc[mt][v], g[o], bb[o]);
    }
  }
}

// ------------------------------------------------- generic GEMM + bn + lrelu
// out[Mtot, ncols] = lrelu(bn(W[Mtot,Ktot] * src[Ktot, ncols]))
__global__ void k_gemm_rm(const float* __restrict__ W, int Mtot, int Ktot,
                          const float* __restrict__ src, int ncols,
                          const float* __restrict__ g,
                          const float* __restrict__ bb,
                          float* __restrict__ out) {
  __shared__ __align__(16) _Float16 sA[SA_HALFS];
  int lane = threadIdx.x & 31;
  int wid = threadIdx.x >> 5;
  int colt = blockIdx.x * (blockDim.x >> 5) + wid;
  int mrow0 = blockIdx.y * 64;
  int col = colt * 16 + (lane & 15);
  int half = lane >> 4;
  int khalf = half << 4;
  v8f acc[4];
  acc[0] = vzero8(); acc[1] = vzero8(); acc[2] = vzero8(); acc[3] = vzero8();
  int Ktiles = (Ktot + 31) >> 5;
  for (int kt = 0; kt < Ktiles; ++kt) {
    __syncthreads();
    fill_sA(sA, W, Mtot, Ktot, mrow0, kt * 32);
    __syncthreads();
    v16h bh, bl;
#pragma unroll
    for (int j = 0; j < 16; ++j) {
      int k = kt * 32 + khalf + j;
      float f = (k < Ktot) ? src[(size_t)k * ncols + col] : 0.f;
      _Float16 h, l;
      splitf(f, h, l);
      bh[j] = h; bl[j] = l;
    }
#pragma unroll
    for (int mt = 0; mt < 4; ++mt) {
      v16h ah, al;
      read_A(sA, mt, lane, ah, al);
      acc[mt] = wmma3(ah, al, bh, bl, acc[mt]);
    }
  }
#pragma unroll
  for (int mt = 0; mt < 4; ++mt) {
#pragma unroll
    for (int v = 0; v < 8; ++v) {
      int o = mrow0 + mt * 16 + half * 8 + v;
      out[(size_t)o * ncols + col] = bnlrelu(acc[mt][v], g[o], bb[o]);
    }
  }
}

// --------------------------------------------------------- final w9 (20x256)
__global__ void k_out9(const float* __restrict__ W,
                       const float* __restrict__ src,  // [256, NPTS]
                       float* __restrict__ out) {      // [NPTS, 20]
  __shared__ __align__(16) _Float16 sA[SA_HALFS];
  int lane = threadIdx.x & 31;
  int wid = threadIdx.x >> 5;
  int colt = blockIdx.x * (blockDim.x >> 5) + wid;
  int col = colt * 16 + (lane & 15);
  int half = lane >> 4;
  int khalf = half << 4;
  v8f acc[2];
  acc[0] = vzero8(); acc[1] = vzero8();
  for (int kt = 0; kt < 8; ++kt) {
    __syncthreads();
    fill_sA(sA, W, 20, 256, 0, kt * 32);
    __syncthreads();
    v16h bh, bl;
#pragma unroll
    for (int j = 0; j < 16; ++j) {
      float f = src[(size_t)(kt * 32 + khalf + j) * NPTS + col];
      _Float16 h, l;
      splitf(f, h, l);
      bh[j] = h; bl[j] = l;
    }
#pragma unroll
    for (int mt = 0; mt < 2; ++mt) {
      v16h ah, al;
      read_A(sA, mt, lane, ah, al);
      acc[mt] = wmma3(ah, al, bh, bl, acc[mt]);
    }
  }
#pragma unroll
  for (int mt = 0; mt < 2; ++mt) {
#pragma unroll
    for (int v = 0; v < 8; ++v) {
      int o = mt * 16 + half * 8 + v;
      if (o < 20) out[(size_t)col * 20 + o] = acc[mt][v];
    }
  }
}

// ---------------------------------------------------------------- reductions
__global__ void k_maxk(const float* __restrict__ src,  // [64, NCOLS]
                       float* __restrict__ dst) {      // [64, NPTS]
  int t = blockIdx.x * blockDim.x + threadIdx.x;
  if (t >= 64 * NPTS) return;
  int o = t / NPTS, n = t % NPTS;
  const float* s = src + (size_t)o * NCOLS + (size_t)n * KNB;
  float m = s[0];
#pragma unroll
  for (int j = 1; j < KNB; ++j) m = fmaxf(m, s[j]);
  dst[t] = m;
}

__global__ void k_colmax(const float* __restrict__ src,  // [1024, NPTS]
                         float* __restrict__ dst) {      // [1024]
  __shared__ float red[256];
  int o = blockIdx.x;
  const float* s = src + (size_t)o * NPTS;
  float m = -3.0e38f;
  for (int n = threadIdx.x; n < NPTS; n += blockDim.x) m = fmaxf(m, s[n]);
  red[threadIdx.x] = m;
  __syncthreads();
  for (int w = 128; w > 0; w >>= 1) {
    if ((int)threadIdx.x < w)
      red[threadIdx.x] = fmaxf(red[threadIdx.x], red[threadIdx.x + w]);
    __syncthreads();
  }
  if (threadIdx.x == 0) dst[o] = red[0];
}

__global__ void k_buildcat(float* __restrict__ dst, const float* __restrict__ x1,
                           const float* __restrict__ x2,
                           const float* __restrict__ x3) {
  int t = blockIdx.x * blockDim.x + threadIdx.x;
  if (t >= 192 * NPTS) return;
  int r = t / NPTS, n = t % NPTS;
  float v;
  if (r < 64) v = x1[r * NPTS + n];
  else if (r < 128) v = x2[(r - 64) * NPTS + n];
  else v = x3[(r - 128) * NPTS + n];
  dst[t] = v;
}

__global__ void k_buildcat7(float* __restrict__ dst,
                            const float* __restrict__ gmax,
                            const float* __restrict__ x1,
                            const float* __restrict__ x2,
                            const float* __restrict__ x3) {
  int t = blockIdx.x * blockDim.x + threadIdx.x;
  if (t >= 1216 * NPTS) return;
  int r = t / NPTS, n = t % NPTS;
  float v;
  if (r < 1024) v = gmax[r];
  else if (r < 1088) v = x1[(r - 1024) * NPTS + n];
  else if (r < 1152) v = x2[(r - 1088) * NPTS + n];
  else v = x3[(r - 1152) * NPTS + n];
  dst[t] = v;
}

// ================================================================== launcher
extern "C" void kernel_launch(void* const* d_in, const int* in_sizes, int n_in,
                              void* d_out, int out_size, void* d_ws,
                              size_t ws_size, hipStream_t stream) {
  (void)in_sizes; (void)n_in; (void)out_size; (void)ws_size;
  const float* coords = (const float*)d_in[0];
  const float* feats  = (const float*)d_in[1];
  const float* w1 = (const float*)d_in[2];
  const float* w2 = (const float*)d_in[3];
  const float* w3 = (const float*)d_in[4];
  const float* w4 = (const float*)d_in[5];
  const float* w5 = (const float*)d_in[6];
  const float* w6 = (const float*)d_in[7];
  const float* w7 = (const float*)d_in[8];
  const float* w8 = (const float*)d_in[9];
  const float* w9 = (const float*)d_in[10];
  const float* g15 = (const float*)d_in[11];
  const float* b15 = (const float*)d_in[12];
  const float* g6 = (const float*)d_in[13];
  const float* b6 = (const float*)d_in[14];
  const float* g7 = (const float*)d_in[15];
  const float* b7 = (const float*)d_in[16];
  const float* g8 = (const float*)d_in[17];
  const float* b8 = (const float*)d_in[18];

  char* ws = (char*)d_ws;
  size_t off = 0;
  auto alloc = [&](size_t bytes) -> void* {
    void* p = ws + off;
    off += (bytes + 255) & ~(size_t)255;
    return p;
  };
  float* x0   = (float*)alloc((size_t)BATCH * 6 * NPTS * 4);
  float* sq   = (float*)alloc((size_t)BATCH * NPTS * 4);
  int*   nbrs = (int*)alloc((size_t)BATCH * NPTS * KNB * 4);
  float* x1   = (float*)alloc((size_t)BATCH * 64 * NPTS * 4);
  float* x2   = (float*)alloc((size_t)BATCH * 64 * NPTS * 4);
  float* x3   = (float*)alloc((size_t)BATCH * 64 * NPTS * 4);
  float* tmp1 = (float*)alloc((size_t)64 * NCOLS * 4);  // per-batch reuse
  float* tmp2 = (float*)alloc((size_t)64 * NCOLS * 4);  // per-batch reuse
  float* gmx  = (float*)alloc((size_t)1024 * 4);        // per-batch reuse

  const dim3 B256(256);
  const int flatB = (BATCH * NPTS + 255) / 256;

  // ---- stage 0: x0, kNN on xyz (full batch)
  k_build_x0<<<flatB, B256, 0, stream>>>(coords, feats, x0);
  k_sq<<<flatB, B256, 0, stream>>>(x0, 3, 6 * NPTS, sq);
  k_knn<3><<<dim3(NPTS / 128, BATCH), dim3(128), 0, stream>>>(x0, 6 * NPTS, sq,
                                                              nbrs);
  // ---- EdgeConv block 1
  for (int b = 0; b < BATCH; ++b) {
    const float* xb = x0 + (size_t)b * 6 * NPTS;
    const int* ib = nbrs + (size_t)b * NPTS * KNB;
    k_edgeconv<<<dim3(NCOLS / 128), B256, 0, stream>>>(w1, 6, xb, ib, g15,
                                                       b15, tmp1);
    k_gemm_rm<<<dim3(NCOLS / 128, 1), B256, 0, stream>>>(
        w2, 64, 64, tmp1, NCOLS, g15 + 64, b15 + 64, tmp2);
    k_maxk<<<dim3(64 * NPTS / 256), B256, 0, stream>>>(tmp2,
                                                       x1 + (size_t)b * 64 * NPTS);
  }
  // ---- EdgeConv block 2 (dynamic kNN in 64-d feature space)
  k_sq<<<flatB, B256, 0, stream>>>(x1, 64, 64 * NPTS, sq);
  k_knn<64><<<dim3(NPTS / 128, BATCH), dim3(128), 0, stream>>>(x1, 64 * NPTS,
                                                               sq, nbrs);
  for (int b = 0; b < BATCH; ++b) {
    const float* xb = x1 + (size_t)b * 64 * NPTS;
    const int* ib = nbrs + (size_t)b * NPTS * KNB;
    k_edgeconv<<<dim3(NCOLS / 128), B256, 0, stream>>>(w3, 64, xb, ib,
                                                       g15 + 128, b15 + 128,
                                                       tmp1);
    k_gemm_rm<<<dim3(NCOLS / 128, 1), B256, 0, stream>>>(
        w4, 64, 64, tmp1, NCOLS, g15 + 192, b15 + 192, tmp2);
    k_maxk<<<dim3(64 * NPTS / 256), B256, 0, stream>>>(tmp2,
                                                       x2 + (size_t)b * 64 * NPTS);
  }
  // ---- EdgeConv block 3
  k_sq<<<flatB, B256, 0, stream>>>(x2, 64, 64 * NPTS, sq);
  k_knn<64><<<dim3(NPTS / 128, BATCH), dim3(128), 0, stream>>>(x2, 64 * NPTS,
                                                               sq, nbrs);
  for (int b = 0; b < BATCH; ++b) {
    const float* xb = x2 + (size_t)b * 64 * NPTS;
    const int* ib = nbrs + (size_t)b * NPTS * KNB;
    k_edgeconv<<<dim3(NCOLS / 128), B256, 0, stream>>>(w5, 64, xb, ib,
                                                       g15 + 256, b15 + 256,
                                                       tmp1);
    k_maxk<<<dim3(64 * NPTS / 256), B256, 0, stream>>>(tmp1,
                                                       x3 + (size_t)b * 64 * NPTS);
  }
  // ---- head: w6 -> global max -> [gmax; x1; x2; x3] -> w7 -> w8 -> w9
  for (int b = 0; b < BATCH; ++b) {
    const float* x1b = x1 + (size_t)b * 64 * NPTS;
    const float* x2b = x2 + (size_t)b * 64 * NPTS;
    const float* x3b = x3 + (size_t)b * 64 * NPTS;
    k_buildcat<<<dim3(192 * NPTS / 256), B256, 0, stream>>>(tmp2, x1b, x2b,
                                                            x3b);
    k_gemm_rm<<<dim3(NPTS / 128, 16), B256, 0, stream>>>(w6, 1024, 192, tmp2,
                                                         NPTS, g6, b6, tmp1);
    k_colmax<<<dim3(1024), B256, 0, stream>>>(tmp1, gmx);
    k_buildcat7<<<dim3(1216 * NPTS / 256), B256, 0, stream>>>(tmp2, gmx, x1b,
                                                              x2b, x3b);
    k_gemm_rm<<<dim3(NPTS / 128, 8), B256, 0, stream>>>(w7, 512, 1216, tmp2,
                                                        NPTS, g7, b7, tmp1);
    k_gemm_rm<<<dim3(NPTS / 128, 4), B256, 0, stream>>>(w8, 256, 512, tmp1,
                                                        NPTS, g8, b8, tmp2);
    k_out9<<<dim3(NPTS / 128), B256, 0, stream>>>(
        w9, tmp2, (float*)d_out + (size_t)b * NPTS * 20);
  }
}
